// DeepGraphSAGE_40578851013002
// MI455X (gfx1250) — matrix operand order repaired and checked by reading, not compiled, not run on hardware
//
#include <hip/hip_runtime.h>
#include <hip/hip_bf16.h>
#include <math.h>

typedef __attribute__((ext_vector_type(16))) _Float16 v16h;
typedef __attribute__((ext_vector_type(8)))  _Float16 v8h;
typedef __attribute__((ext_vector_type(8)))  float    v8f;

#define N_NODES 50000
#define N_EDGES 400000
#define MPAD    50048          // multiple of 128
#define HID     512
#define OUTF    121
#define OUTP    128
#define EPS     1e-5f

#define TILE_M 128
#define TILE_N 128
#define TILE_K 32
#define LDS_STRIDE 40          // halfs; 80B rows: 16B-aligned, bank-spread

__device__ __forceinline__ void atomicAddF32(float* p, float v) {
    __hip_atomic_fetch_add(p, v, __ATOMIC_RELAXED, __HIP_MEMORY_SCOPE_AGENT);
}

// ---------------------------------------------------------------- utilities
__global__ void zero_f32(float* p, size_t n) {
    size_t i = (size_t)blockIdx.x * blockDim.x + threadIdx.x;
    size_t stride = (size_t)gridDim.x * blockDim.x;
    for (; i < n; i += stride) p[i] = 0.0f;
}

__global__ void degree_kernel(const long long* __restrict__ dst, int nEdges,
                              float* __restrict__ deg) {
    int e = blockIdx.x * blockDim.x + threadIdx.x;
    if (e < nEdges) atomicAddF32(&deg[(int)dst[e]], 1.0f);
}

// one block per edge; threads stride the feature dim
__global__ void scatter_add(const float* __restrict__ feat,
                            const long long* __restrict__ src,
                            const long long* __restrict__ dst,
                            float* __restrict__ agg,
                            int F, int ldf, int lda) {
    int e = blockIdx.x;
    int s = (int)src[e];
    int d = (int)dst[e];
    for (int f = threadIdx.x; f < F; f += blockDim.x)
        atomicAddF32(&agg[(size_t)d * lda + f], feat[(size_t)s * ldf + f]);
}

// ------------------------------------------------- f16 packing (GEMM inputs)
// layer 1: Xc row = [ agg/deg (50, pad 64) | x (50, pad 64) ]  -> K = 128
__global__ void pack_l1(const float* __restrict__ agg, const float* __restrict__ deg,
                        const float* __restrict__ x, _Float16* __restrict__ Xc,
                        int nReal) {
    int row = blockIdx.x;
    int t   = threadIdx.x;              // 0..127
    float v = 0.0f;
    if (row < nReal) {
        if (t < 50)                 v = agg[(size_t)row * 64 + t] * (1.0f / fmaxf(deg[row], 1.0f));
        else if (t >= 64 && t < 114) v = x[(size_t)row * 50 + (t - 64)];
    }
    Xc[(size_t)row * 128 + t] = (_Float16)v;
}

// layers 2+: Xc row = [ agg/deg (512) | y (512) ]  -> K = 1024
__global__ void pack_h(const float* __restrict__ agg, const float* __restrict__ deg,
                       const float* __restrict__ y, _Float16* __restrict__ Xc,
                       int nReal) {
    int row = blockIdx.x;
    int col = blockIdx.y * blockDim.x + threadIdx.x;   // 0..1023
    float v = 0.0f;
    if (row < nReal) {
        if (col < HID) v = agg[(size_t)row * HID + col] * (1.0f / fmaxf(deg[row], 1.0f));
        else           v = y[(size_t)row * HID + (col - HID)];
    }
    Xc[(size_t)row * 1024 + col] = (_Float16)v;
}

// pack [Wl; Wr] (f32, row-major KinReal x NoutReal each) into B-fragment-native
// f16 layout: element (k, n) -> Wp[((k/32)*NoutPad + n)*32 + (k%32)]
__global__ void pack_w(const float* __restrict__ Wl, const float* __restrict__ Wr,
                       _Float16* __restrict__ Wp,
                       int KinReal, int KinPad, int NoutReal, int NoutPad) {
    int idx = blockIdx.x * blockDim.x + threadIdx.x;
    int total = 2 * KinPad * NoutPad;
    if (idx >= total) return;
    int k = idx / NoutPad;
    int n = idx - k * NoutPad;
    float v = 0.0f;
    if (n < NoutReal) {
        if (k < KinPad) { if (k < KinReal) v = Wl[(size_t)k * NoutReal + n]; }
        else { int kk = k - KinPad; if (kk < KinReal) v = Wr[(size_t)kk * NoutReal + n]; }
    }
    Wp[((size_t)(k >> 5) * NoutPad + n) * 32 + (k & 31)] = (_Float16)v;
}

// --------------------------------------------------------------- WMMA GEMM
// C[M x Nreal] (f32, ld=ldc) = Xc[MPAD x Kc](f16) * Wp(packed f16) + bias
__global__ __launch_bounds__(256)
void gemm_wmma(const _Float16* __restrict__ Xc, int Kc,
               const _Float16* __restrict__ Wp, int Noutp,
               const float* __restrict__ bias, int Nreal,
               float* __restrict__ C, int ldc, int Mreal) {
    __shared__ _Float16 sA[TILE_M * LDS_STRIDE];

    const int tid    = threadIdx.x;
    const int lane   = tid & 31;
    const int wave   = tid >> 5;        // 0..7
    const int waveM  = wave & 3;        // 4 waves x 32 rows
    const int waveN  = wave >> 2;       // 2 waves x 64 cols
    const int blockM = blockIdx.x * TILE_M;
    const int blockN = blockIdx.y * TILE_N;

    const int lane15   = lane & 15;
    const int laneHalf = lane >> 4;
    const int kA0 = laneHalf * 8;       // A frag K base (ISA 16-bit A layout)
    const int kB0 = laneHalf * 8;       // B frag K base (same K order as A)

    v8f acc[2][4];
#pragma unroll
    for (int mt = 0; mt < 2; ++mt)
#pragma unroll
        for (int nt = 0; nt < 4; ++nt) acc[mt][nt] = (v8f)(0.0f);

    const int nKB = Kc / TILE_K;
    for (int kb = 0; kb < nKB; ++kb) {
        // cooperative A tile load: 512 x 16B units, 2 per thread
#pragma unroll
        for (int u = 0; u < 2; ++u) {
            int unit = tid + u * 256;
            int row  = unit >> 2;
            int c8   = (unit & 3) * 8;
            const v8h* srcp = (const v8h*)(Xc + (size_t)(blockM + row) * Kc + kb * TILE_K + c8);
            *(v8h*)(sA + row * LDS_STRIDE + c8) = *srcp;
            if (kb + 1 < nKB)   // global_prefetch next K-block
                __builtin_prefetch(Xc + (size_t)(blockM + row) * Kc + (kb + 1) * TILE_K + c8, 0, 1);
        }
        __syncthreads();

        // B fragments: two contiguous 16B loads per 16-col tile (packed layout)
        v16h bfrag[4];
#pragma unroll
        for (int nt = 0; nt < 4; ++nt) {
            int ncol = blockN + waveN * 64 + nt * 16 + lane15;
            const _Float16* bp = Wp + ((size_t)kb * Noutp + ncol) * 32 + kB0;
            v8h blo = *(const v8h*)bp;
            v8h bhi = *(const v8h*)(bp + 16);
            v16h b;
#pragma unroll
            for (int i = 0; i < 8; ++i) { b[i] = blo[i]; b[i + 8] = bhi[i]; }
            bfrag[nt] = b;
        }

#pragma unroll
        for (int mt = 0; mt < 2; ++mt) {
            const _Float16* ap = sA + (waveM * 32 + mt * 16 + lane15) * LDS_STRIDE + kA0;
            v8h alo = *(const v8h*)ap;         // K 0-7 / 8-15
            v8h ahi = *(const v8h*)(ap + 16);  // K 16-23 / 24-31
            v16h a;
#pragma unroll
            for (int i = 0; i < 8; ++i) { a[i] = alo[i]; a[i + 8] = ahi[i]; }
#pragma unroll
            for (int nt = 0; nt < 4; ++nt)
                acc[mt][nt] = __builtin_amdgcn_wmma_f32_16x16x32_f16(
                    false, a, false, bfrag[nt], (short)0, acc[mt][nt], false, false);
        }
        __syncthreads();
    }

    // epilogue: C/D layout — VGPR i: lanes0-15 M=i, lanes16-31 M=i+8; N=lane%16
#pragma unroll
    for (int mt = 0; mt < 2; ++mt)
#pragma unroll
        for (int nt = 0; nt < 4; ++nt) {
            int col = blockN + waveN * 64 + nt * 16 + lane15;
            if (col >= Nreal) continue;
            float bv = bias[col];
#pragma unroll
            for (int i = 0; i < 8; ++i) {
                int row = blockM + waveM * 32 + mt * 16 + i + laneHalf * 8;
                if (row < Mreal) C[(size_t)row * ldc + col] = acc[mt][nt][i] + bv;
            }
        }
}

// ------------------------------------------------------------- batch norm
#define BN_ROWS 128
__global__ void bn_stats(const float* __restrict__ h, int nRows,
                         float* __restrict__ sums, float* __restrict__ sqs) {
    int col = threadIdx.x;                       // 256 threads -> 2 cols each
    int r0 = blockIdx.x * BN_ROWS;
    int rend = min(r0 + BN_ROWS, nRows);
    float s0 = 0.f, s1 = 0.f, q0 = 0.f, q1 = 0.f;
    for (int r = r0; r < rend; ++r) {
        float a = h[(size_t)r * HID + col];
        float b = h[(size_t)r * HID + col + 256];
        s0 += a; q0 += a * a; s1 += b; q1 += b * b;
    }
    atomicAddF32(&sums[col], s0);
    atomicAddF32(&sums[col + 256], s1);
    atomicAddF32(&sqs[col], q0);
    atomicAddF32(&sqs[col + 256], q1);
}

__global__ void bn_apply_relu(float* __restrict__ h,
                              const float* __restrict__ sums, const float* __restrict__ sqs,
                              const float* __restrict__ g, const float* __restrict__ b,
                              int nRows, float invN) {
    size_t idx = (size_t)blockIdx.x * blockDim.x + threadIdx.x;
    if (idx >= (size_t)nRows * HID) return;
    int col = (int)(idx & (HID - 1));
    float mu  = sums[col] * invN;
    float var = sqs[col] * invN - mu * mu;
    float v = (h[idx] - mu) * rsqrtf(var + EPS) * g[col] + b[col];
    h[idx] = fmaxf(v, 0.0f);
}

// ---------------------------------------------------------------- launcher
extern "C" void kernel_launch(void* const* d_in, const int* in_sizes, int n_in,
                              void* d_out, int out_size, void* d_ws, size_t ws_size,
                              hipStream_t stream) {
    const float*     x   = (const float*)d_in[0];
    const long long* ei  = (const long long*)d_in[1];
    const long long* src = ei;
    const long long* dst = ei + N_EDGES;

    const float* Wl[4] = { (const float*)d_in[2], (const float*)d_in[5], (const float*)d_in[8],  (const float*)d_in[11] };
    const float* bl[4] = { (const float*)d_in[3], (const float*)d_in[6], (const float*)d_in[9],  (const float*)d_in[12] };
    const float* Wr[4] = { (const float*)d_in[4], (const float*)d_in[7], (const float*)d_in[10], (const float*)d_in[13] };
    const float* gn[3] = { (const float*)d_in[14], (const float*)d_in[16], (const float*)d_in[18] };
    const float* bn[3] = { (const float*)d_in[15], (const float*)d_in[17], (const float*)d_in[19] };

    // workspace carve-up (256B aligned)
    size_t off = 0;
    auto carve = [&](size_t bytes) {
        void* p = (char*)d_ws + off;
        off += (bytes + 255) & ~(size_t)255;
        return p;
    };
    float*     deg  = (float*)carve((size_t)MPAD * 4);
    float*     agg  = (float*)carve((size_t)MPAD * HID * 4);
    float*     fbuf = (float*)carve((size_t)MPAD * HID * 4);   // GEMM out / BN in-place / scatter src
    _Float16*  Xc   = (_Float16*)carve((size_t)MPAD * 1024 * 2);
    _Float16*  Wp   = (_Float16*)carve((size_t)1024 * HID * 2);
    float*     sums = (float*)carve(1024 * 4);                 // [512 sums | 512 sqsums]

    const int ZB = 1024;  // blocks for grid-stride zero kernels

    // degrees (reused by every layer)
    zero_f32<<<ZB, 256, 0, stream>>>(deg, (size_t)MPAD);
    degree_kernel<<<(N_EDGES + 255) / 256, 256, 0, stream>>>(dst, N_EDGES, deg);

    // ----- layer 1: in 50 (pad 64), K=128, out 512
    zero_f32<<<ZB, 256, 0, stream>>>(agg, (size_t)MPAD * 64);
    scatter_add<<<N_EDGES, 64, 0, stream>>>(x, src, dst, agg, 50, 50, 64);
    pack_l1<<<MPAD, 128, 0, stream>>>(agg, deg, x, Xc, N_NODES);
    pack_w<<<(2 * 64 * HID + 255) / 256, 256, 0, stream>>>(Wl[0], Wr[0], Wp, 50, 64, HID, HID);
    gemm_wmma<<<dim3(MPAD / TILE_M, HID / TILE_N), 256, 0, stream>>>(
        Xc, 128, Wp, HID, bl[0], HID, fbuf, HID, N_NODES);
    zero_f32<<<4, 256, 0, stream>>>(sums, 1024);
    bn_stats<<<(N_NODES + BN_ROWS - 1) / BN_ROWS, 256, 0, stream>>>(fbuf, N_NODES, sums, sums + 512);
    bn_apply_relu<<<((size_t)N_NODES * HID + 255) / 256, 256, 0, stream>>>(
        fbuf, sums, sums + 512, gn[0], bn[0], N_NODES, 1.0f / N_NODES);

    // ----- layers 2 and 3: 512 -> 512, K=1024, BN+ReLU
    for (int L = 1; L <= 2; ++L) {
        zero_f32<<<ZB, 256, 0, stream>>>(agg, (size_t)MPAD * HID);
        scatter_add<<<N_EDGES, 128, 0, stream>>>(fbuf, src, dst, agg, HID, HID, HID);
        pack_h<<<dim3(MPAD, 4), 256, 0, stream>>>(agg, deg, fbuf, Xc, N_NODES);
        pack_w<<<(2 * HID * HID + 255) / 256, 256, 0, stream>>>(Wl[L], Wr[L], Wp, HID, HID, HID, HID);
        gemm_wmma<<<dim3(MPAD / TILE_M, HID / TILE_N), 256, 0, stream>>>(
            Xc, 1024, Wp, HID, bl[L], HID, fbuf, HID, N_NODES);
        zero_f32<<<4, 256, 0, stream>>>(sums, 1024);
        bn_stats<<<(N_NODES + BN_ROWS - 1) / BN_ROWS, 256, 0, stream>>>(fbuf, N_NODES, sums, sums + 512);
        bn_apply_relu<<<((size_t)N_NODES * HID + 255) / 256, 256, 0, stream>>>(
            fbuf, sums, sums + 512, gn[L], bn[L], N_NODES, 1.0f / N_NODES);
    }

    // ----- layer 4: 512 -> 121 (pad 128), K=1024, no BN
    zero_f32<<<ZB, 256, 0, stream>>>(agg, (size_t)MPAD * HID);
    scatter_add<<<N_EDGES, 128, 0, stream>>>(fbuf, src, dst, agg, HID, HID, HID);
    pack_h<<<dim3(MPAD, 4), 256, 0, stream>>>(agg, deg, fbuf, Xc, N_NODES);
    pack_w<<<(2 * HID * OUTP + 255) / 256, 256, 0, stream>>>(Wl[3], Wr[3], Wp, HID, HID, OUTF, OUTP);
    gemm_wmma<<<dim3(MPAD / TILE_M, OUTP / TILE_N), 256, 0, stream>>>(
        Xc, 1024, Wp, OUTP, bl[3], OUTF, (float*)d_out, OUTF, N_NODES);

    (void)in_sizes; (void)n_in; (void)out_size; (void)ws_size;
}
